// AdvancedMambaAMT_39324720562847
// MI455X (gfx1250) — matrix operand order repaired and checked by reading, not compiled, not run on hardware
//
#include <hip/hip_runtime.h>
#include <hip/hip_bf16.h>

// ---------------------------------------------------------------------------
// AdvancedMambaAMT forward for gfx1250 (MI455X).
// All dense GEMMs + attention GEMMs use v_wmma_f32_16x16x32_f16 (f16 in, f32
// accum). Fragment layouts used here (A and B share one k order):
//   A (16x32 f16): lane L<16 -> row L, K = 0..15; lane L+16 -> row L,
//                  K = 16..31  (contiguous halves, 2 f16 / VGPR)
//   B (32x16 f16): lane c<16 -> col c, K = 0..15 sequential; lane c+16 ->
//                  col c, K = 16..31   (matches the sparse-B table pattern)
//   C/D (16x16 f32): lane c<16 -> col c, VGPR r -> row r; lane c+16 -> col c,
//                  VGPR r -> row r+8
// ---------------------------------------------------------------------------

typedef __attribute__((ext_vector_type(16))) _Float16 v16h;
typedef __attribute__((ext_vector_type(8)))  float    v8f;

#define B_   32
#define N_   512
#define NH_  4
#define HD_  128
#define M_   (B_ * N_)   // 16384 rows

__device__ __forceinline__ v8f v8f_zero() {
  v8f v = {0.f,0.f,0.f,0.f,0.f,0.f,0.f,0.f};
  return v;
}

__device__ __forceinline__ v8f wmma_f16(v16h a, v16h b, v8f c) {
  return __builtin_amdgcn_wmma_f32_16x16x32_f16(false, a, false, b,
                                                (short)0, c, false, false);
}

// A-fragment (16x32) from row-major fp32, K contiguous (ld = row stride).
__device__ __forceinline__ v16h frag_a_global(const float* __restrict__ p, int ld) {
  const int lane = threadIdx.x & 31;
  const float* r = p + (size_t)(lane & 15) * ld + ((lane >> 4) << 4);
  float4 a0 = reinterpret_cast<const float4*>(r)[0];
  float4 a1 = reinterpret_cast<const float4*>(r)[1];
  float4 c0 = reinterpret_cast<const float4*>(r + 8)[0];
  float4 c1 = reinterpret_cast<const float4*>(r + 8)[1];
  v16h f;
  f[0]=(_Float16)a0.x; f[1]=(_Float16)a0.y; f[2]=(_Float16)a0.z; f[3]=(_Float16)a0.w;
  f[4]=(_Float16)a1.x; f[5]=(_Float16)a1.y; f[6]=(_Float16)a1.z; f[7]=(_Float16)a1.w;
  f[8]=(_Float16)c0.x; f[9]=(_Float16)c0.y; f[10]=(_Float16)c0.z; f[11]=(_Float16)c0.w;
  f[12]=(_Float16)c1.x; f[13]=(_Float16)c1.y; f[14]=(_Float16)c1.z; f[15]=(_Float16)c1.w;
  return f;
}

// A-fragment from f16 LDS tile (row stride ld).
__device__ __forceinline__ v16h frag_a_lds(const _Float16* p, int ld) {
  const int lane = threadIdx.x & 31;
  const _Float16* r = p + (lane & 15) * ld + ((lane >> 4) << 4);
  v16h f;
#pragma unroll
  for (int j = 0; j < 8; ++j) f[j] = r[j];
#pragma unroll
  for (int j = 0; j < 8; ++j) f[8 + j] = r[8 + j];
  return f;
}

// B-fragment (32x16) from f16 LDS where B source is row-major (N x K):
// lane = output column n, 16 sequential K per lane half.
__device__ __forceinline__ v16h frag_b_lds_nk(const _Float16* p, int ld) {
  const int lane = threadIdx.x & 31;
  const _Float16* r = p + (lane & 15) * ld + ((lane >> 4) << 4);
  v16h f;
#pragma unroll
  for (int j = 0; j < 16; ++j) f[j] = r[j];
  return f;
}

// B-fragment (32x16) from f16 LDS where B source is row-major (K x N):
// lane = column n, strided walk down K.
__device__ __forceinline__ v16h frag_b_lds_kn(const _Float16* p, int ld) {
  const int lane = threadIdx.x & 31;
  const int n = lane & 15;
  const int k0 = (lane >> 4) << 4;
  v16h f;
#pragma unroll
  for (int j = 0; j < 16; ++j) f[j] = p[(k0 + j) * ld + n];
  return f;
}

// ---------------------------------------------------------------------------
// Generic WMMA GEMM: Y[M,N] = act(A[M,K] @ W[N,K]^T + bias). act: 0=id, 1=sigmoid
// Block 256 threads (8 waves) -> 128x64 tile; K step 32; W staged via LDS.
// ---------------------------------------------------------------------------
__global__ __launch_bounds__(256) void gemm_wmma_nt(
    const float* __restrict__ A, const float* __restrict__ W,
    const float* __restrict__ bias, float* __restrict__ Y,
    int M, int N, int K, int act)
{
  __shared__ _Float16 lb[64 * 40];          // 64 n-rows x 32 k (+8 pad)
  const int tid = threadIdx.x;
  const int wv  = tid >> 5;
  const int n0  = blockIdx.x * 64;
  const int m0  = blockIdx.y * 128 + wv * 16;

  v8f acc[4];
#pragma unroll
  for (int i = 0; i < 4; ++i) acc[i] = v8f_zero();

  const int srow = tid >> 2;                // 0..63
  const int scol = (tid & 3) << 3;          // 0,8,16,24

  for (int k0 = 0; k0 < K; k0 += 32) {
    __syncthreads();
    {
      const float* src = W + (size_t)(n0 + srow) * K + k0 + scol;
      float4 w0 = reinterpret_cast<const float4*>(src)[0];
      float4 w1 = reinterpret_cast<const float4*>(src)[1];
      _Float16* dst = lb + srow * 40 + scol;
      dst[0]=(_Float16)w0.x; dst[1]=(_Float16)w0.y; dst[2]=(_Float16)w0.z; dst[3]=(_Float16)w0.w;
      dst[4]=(_Float16)w1.x; dst[5]=(_Float16)w1.y; dst[6]=(_Float16)w1.z; dst[7]=(_Float16)w1.w;
    }
    __syncthreads();
    v16h af = frag_a_global(A + (size_t)m0 * K + k0, K);
#pragma unroll
    for (int nt = 0; nt < 4; ++nt) {
      v16h bf = frag_b_lds_nk(lb + nt * 16 * 40, 40);
      acc[nt] = wmma_f16(af, bf, acc[nt]);
    }
  }

  const int lane = tid & 31;
  const int col  = lane & 15;
  const int rb   = (lane >> 4) << 3;
#pragma unroll
  for (int nt = 0; nt < 4; ++nt) {
    const int n = n0 + nt * 16 + col;
    float bv = bias ? bias[n] : 0.0f;
#pragma unroll
    for (int r = 0; r < 8; ++r) {
      float yv = acc[nt][r] + bv;
      if (act == 1) yv = 1.0f / (1.0f + expf(-yv));
      Y[(size_t)(m0 + rb + r) * N + n] = yv;
    }
  }
}

// ---------------------------------------------------------------------------
// LayerNorm (optionally + exact GELU). One block per row. In-place safe.
// ---------------------------------------------------------------------------
__global__ __launch_bounds__(256) void layernorm_kernel(
    const float* __restrict__ x, const float* __restrict__ g,
    const float* __restrict__ bsh, float* __restrict__ y, int C, int do_gelu)
{
  __shared__ float r1[256], r2[256];
  const int row = blockIdx.x, tid = threadIdx.x;
  const float* xr = x + (size_t)row * C;
  float s = 0.f, s2 = 0.f;
  for (int c = tid; c < C; c += 256) { float v = xr[c]; s += v; s2 += v * v; }
  r1[tid] = s; r2[tid] = s2; __syncthreads();
  for (int off = 128; off > 0; off >>= 1) {
    if (tid < off) { r1[tid] += r1[tid + off]; r2[tid] += r2[tid + off]; }
    __syncthreads();
  }
  const float mean = r1[0] / (float)C;
  const float var  = r2[0] / (float)C - mean * mean;
  const float rstd = rsqrtf(var + 1e-5f);
  float* yr = y + (size_t)row * C;
  for (int c = tid; c < C; c += 256) {
    float v = (xr[c] - mean) * rstd * g[c] + bsh[c];
    if (do_gelu) v = 0.5f * v * (1.0f + erff(v * 0.70710678118654752f));
    yr[c] = v;
  }
}

// ---------------------------------------------------------------------------
// Combine + LayerNorm over C=512.
//   mode 0: t = gamma[0]*p0 + p1 + 0.1*p2   (FE residual mix)
//   mode 1: t = p0 + p1*p2                  (gated residual)
// ---------------------------------------------------------------------------
__global__ __launch_bounds__(256) void combine_ln_kernel(
    int mode, const float* __restrict__ p0, const float* __restrict__ p1,
    const float* __restrict__ p2, const float* __restrict__ gamma,
    const float* __restrict__ g, const float* __restrict__ bsh,
    float* __restrict__ y)
{
  __shared__ float r1[256], r2[256];
  const int row = blockIdx.x, tid = threadIdx.x;
  const size_t base = (size_t)row * 512;
  const float gm = (mode == 0) ? gamma[0] : 0.0f;
  float vv[2];
#pragma unroll
  for (int j = 0; j < 2; ++j) {
    const size_t i = base + tid + j * 256;
    vv[j] = (mode == 0) ? (gm * p0[i] + p1[i] + 0.1f * p2[i])
                        : (p0[i] + p1[i] * p2[i]);
  }
  float s = vv[0] + vv[1], s2 = vv[0] * vv[0] + vv[1] * vv[1];
  r1[tid] = s; r2[tid] = s2; __syncthreads();
  for (int off = 128; off > 0; off >>= 1) {
    if (tid < off) { r1[tid] += r1[tid + off]; r2[tid] += r2[tid + off]; }
    __syncthreads();
  }
  const float mean = r1[0] * (1.0f / 512.0f);
  const float var  = r2[0] * (1.0f / 512.0f) - mean * mean;
  const float rstd = rsqrtf(var + 1e-5f);
#pragma unroll
  for (int j = 0; j < 2; ++j) {
    const int c = tid + j * 256;
    y[base + c] = (vv[j] - mean) * rstd * g[c] + bsh[c];
  }
}

// ---------------------------------------------------------------------------
// Flash-style causal attention with relative bias, WMMA for QK^T and PV.
// Grid: (8 q-tiles of 64 rows, B*NH). Block: 128 threads = 4 waves x 16 rows.
// ---------------------------------------------------------------------------
#define LDH 136
__global__ __launch_bounds__(128) void attn_kernel(
    const float* __restrict__ q, const float* __restrict__ k,
    const float* __restrict__ v, const float* __restrict__ rel_bias,
    float* __restrict__ out)
{
  __shared__ _Float16 lk[32 * LDH];
  __shared__ _Float16 lv[32 * LDH];
  __shared__ _Float16 pb[4 * 16 * 32];

  const int tid  = threadIdx.x;
  const int wv   = tid >> 5;
  const int lane = tid & 31;
  const int bh   = blockIdx.y;
  const int bb   = bh >> 2;
  const int h    = bh & 3;
  const int qb   = blockIdx.x * 64;
  const int q0   = qb + wv * 16;
  const size_t base  = ((size_t)bb * N_) * 512 + (size_t)h * HD_;
  const size_t bbase = (size_t)h * N_ * N_;
  const float scale = 0.08838834764831845f;   // 1/sqrt(128)

  v16h qf[4];
#pragma unroll
  for (int kc = 0; kc < 4; ++kc)
    qf[kc] = frag_a_global(q + base + (size_t)q0 * 512 + kc * 32, 512);

  v8f o[8];
#pragma unroll
  for (int i = 0; i < 8; ++i) o[i] = v8f_zero();
  float rmax[8], rsum[8];
#pragma unroll
  for (int i = 0; i < 8; ++i) { rmax[i] = -1e30f; rsum[i] = 0.f; }

  const int srow = tid >> 2;              // 0..31
  const int scol = (tid & 3) << 5;        // 0,32,64,96
  const int col  = lane & 15;
  const int rb   = (lane >> 4) << 3;
  _Float16* mypb = pb + wv * (16 * 32);

  for (int j0 = 0; j0 < qb + 64; j0 += 32) {
    __syncthreads();
    {
      const float* ks = k + base + (size_t)(j0 + srow) * 512 + scol;
      const float* vs = v + base + (size_t)(j0 + srow) * 512 + scol;
      _Float16* dk = lk + srow * LDH + scol;
      _Float16* dv = lv + srow * LDH + scol;
#pragma unroll
      for (int j = 0; j < 8; ++j) {
        float4 kk = reinterpret_cast<const float4*>(ks)[j];
        float4 vvv = reinterpret_cast<const float4*>(vs)[j];
        dk[4*j+0]=(_Float16)kk.x;  dk[4*j+1]=(_Float16)kk.y;
        dk[4*j+2]=(_Float16)kk.z;  dk[4*j+3]=(_Float16)kk.w;
        dv[4*j+0]=(_Float16)vvv.x; dv[4*j+1]=(_Float16)vvv.y;
        dv[4*j+2]=(_Float16)vvv.z; dv[4*j+3]=(_Float16)vvv.w;
      }
    }
    __syncthreads();
    if (j0 < q0 + 16) {           // wave-uniform: EXEC stays all-ones
      v8f s0 = v8f_zero(), s1 = v8f_zero();
#pragma unroll
      for (int kc = 0; kc < 4; ++kc) {
        v16h bf0 = frag_b_lds_nk(lk + kc * 32, LDH);
        s0 = wmma_f16(qf[kc], bf0, s0);
        v16h bf1 = frag_b_lds_nk(lk + 16 * LDH + kc * 32, LDH);
        s1 = wmma_f16(qf[kc], bf1, s1);
      }
#pragma unroll
      for (int r = 0; r < 8; ++r) {
        const int row = q0 + rb + r;
        float x0 = s0[r] * scale + rel_bias[bbase + (size_t)row * 512 + (j0 + col)];
        float x1 = s1[r] * scale + rel_bias[bbase + (size_t)row * 512 + (j0 + 16 + col)];
        if (j0 + col      > row) x0 = -1e30f;
        if (j0 + 16 + col > row) x1 = -1e30f;
        float mx = fmaxf(x0, x1);
        mx = fmaxf(mx, __shfl_xor(mx, 1, 32));
        mx = fmaxf(mx, __shfl_xor(mx, 2, 32));
        mx = fmaxf(mx, __shfl_xor(mx, 4, 32));
        mx = fmaxf(mx, __shfl_xor(mx, 8, 32));
        const float nm   = fmaxf(rmax[r], mx);
        const float corr = expf(rmax[r] - nm);
        const float p0v  = expf(x0 - nm);
        const float p1v  = expf(x1 - nm);
        float ps = p0v + p1v;
        ps += __shfl_xor(ps, 1, 32);
        ps += __shfl_xor(ps, 2, 32);
        ps += __shfl_xor(ps, 4, 32);
        ps += __shfl_xor(ps, 8, 32);
        rsum[r] = rsum[r] * corr + ps;
        rmax[r] = nm;
#pragma unroll
        for (int dt = 0; dt < 8; ++dt) o[dt][r] *= corr;
        mypb[(rb + r) * 32 + col]      = (_Float16)p0v;
        mypb[(rb + r) * 32 + 16 + col] = (_Float16)p1v;
      }
      // per-wave LDS round-trip to re-layout P (D layout) into an A fragment
      asm volatile("s_wait_dscnt 0x0" ::: "memory");
      v16h pf = frag_a_lds(mypb, 32);
#pragma unroll
      for (int dt = 0; dt < 8; ++dt) {
        v16h vf = frag_b_lds_kn(lv + dt * 16, LDH);
        o[dt] = wmma_f16(pf, vf, o[dt]);
      }
    }
  }
#pragma unroll
  for (int dt = 0; dt < 8; ++dt)
#pragma unroll
    for (int r = 0; r < 8; ++r)
      out[base + (size_t)(q0 + rb + r) * 512 + dt * 16 + col] = o[dt][r] / rsum[r];
}

// ---------------------------------------------------------------------------
// Depthwise temporal conv (k=3, SAME) added into att_out.
// ---------------------------------------------------------------------------
__global__ __launch_bounds__(256) void conv_add_kernel(
    const float* __restrict__ v, const float* __restrict__ w,
    const float* __restrict__ cb, float* __restrict__ att)
{
  const size_t i = (size_t)blockIdx.x * 256 + threadIdx.x;   // < B*N*512
  const int c = (int)(i & 511);
  const int t = (int)((i >> 9) & 511);
  float acc = cb[c] + w[c * 3 + 1] * v[i];
  if (t > 0)   acc += w[c * 3 + 0] * v[i - 512];
  if (t < 511) acc += w[c * 3 + 2] * v[i + 512];
  att[i] += acc;
}

// ---------------------------------------------------------------------------
// AMT causal linear attention scan. One block per (b,h), 256 threads.
// Thread (e = tid>>1, half = tid&1) owns S[half*64 .. +63][e] in registers.
// ---------------------------------------------------------------------------
__global__ __launch_bounds__(256) void amt_scan_kernel(
    const float* __restrict__ q, const float* __restrict__ k,
    const float* __restrict__ v, float* __restrict__ amt)
{
  __shared__ float kf[128], qf[128], z[128];
  const int bh = blockIdx.x;
  const int bb = bh >> 2, h = bh & 3;
  const size_t base = ((size_t)bb * N_) * 512 + (size_t)h * HD_;
  const int tid = threadIdx.x;
  const int e = tid >> 1, half = tid & 1;
  const int d0 = half * 64;

  if (tid < 128) z[tid] = 0.f;
  float S[64];
#pragma unroll
  for (int i = 0; i < 64; ++i) S[i] = 0.f;

  for (int t = 0; t < 512; ++t) {
    __syncthreads();
    if (tid < 128) {
      float kv = k[base + (size_t)t * 512 + tid];
      float qv = q[base + (size_t)t * 512 + tid];
      kv = (kv > 0.f) ? kv + 1.f : expf(kv);   // elu+1
      qv = (qv > 0.f) ? qv + 1.f : expf(qv);
      kf[tid] = kv; qf[tid] = qv; z[tid] += kv;
    }
    __syncthreads();
    const float ve = v[base + (size_t)t * 512 + e];
    float num = 0.f, den = 0.f;
#pragma unroll
    for (int i = 0; i < 64; ++i) {
      S[i] = fmaf(kf[d0 + i], ve, S[i]);
      num  = fmaf(qf[d0 + i], S[i], num);
      den  = fmaf(qf[d0 + i], z[d0 + i], den);
    }
    num += __shfl_xor(num, 1, 32);
    den += __shfl_xor(den, 1, 32);
    if (half == 0) amt[base + (size_t)t * 512 + e] = num / (den + 1e-6f);
  }
}

// ---------------------------------------------------------------------------
// Elementwise helpers.
// ---------------------------------------------------------------------------
__global__ __launch_bounds__(256) void concat_kernel(
    const float* __restrict__ a, const float* __restrict__ b, float* __restrict__ y)
{
  const size_t i = (size_t)blockIdx.x * 256 + threadIdx.x;   // < M*1024
  const size_t row = i >> 10;
  const int c = (int)(i & 1023);
  y[i] = (c < 512) ? a[row * 512 + c] : b[row * 512 + (c - 512)];
}

__global__ __launch_bounds__(256) void fuse_kernel(
    const float* __restrict__ g, const float* __restrict__ amt,
    const float* __restrict__ att, float* __restrict__ y)
{
  const size_t i = (size_t)blockIdx.x * 256 + threadIdx.x;   // < M*512
  const float gg = g[i];
  y[i] = gg * amt[i] + (1.0f - gg) * att[i];
}

// ---------------------------------------------------------------------------
// Host driver.
// ---------------------------------------------------------------------------
extern "C" void kernel_launch(void* const* d_in, const int* in_sizes, int n_in,
                              void* d_out, int out_size, void* d_ws, size_t ws_size,
                              hipStream_t stream)
{
  (void)in_sizes; (void)n_in; (void)out_size; (void)ws_size;

  const float* x      = (const float*)d_in[0];
  const float* fe_w1  = (const float*)d_in[1];
  const float* fe_b1  = (const float*)d_in[2];
  const float* fe_w2  = (const float*)d_in[3];
  const float* fe_b2  = (const float*)d_in[4];
  const float* fe_wsc = (const float*)d_in[5];
  const float* fe_bsc = (const float*)d_in[6];
  const float* ln1_g  = (const float*)d_in[7];
  const float* ln1_b  = (const float*)d_in[8];
  const float* ln2_g  = (const float*)d_in[9];
  const float* ln2_b  = (const float*)d_in[10];
  const float* gamma  = (const float*)d_in[11];
  const float* wq     = (const float*)d_in[12];
  const float* wk     = (const float*)d_in[13];
  const float* wv     = (const float*)d_in[14];
  const float* conv_w = (const float*)d_in[15];
  const float* conv_b = (const float*)d_in[16];
  const float* relb   = (const float*)d_in[17];
  const float* gate_w = (const float*)d_in[18];
  const float* gate_b = (const float*)d_in[19];
  const float* rg_w   = (const float*)d_in[20];
  const float* rg_b   = (const float*)d_in[21];
  const float* norm_g = (const float*)d_in[22];
  const float* norm_b = (const float*)d_in[23];
  const float* fc1_w  = (const float*)d_in[24];
  const float* fc1_b  = (const float*)d_in[25];
  const float* fcln_g = (const float*)d_in[26];
  const float* fcln_b = (const float*)d_in[27];
  const float* fc2_w  = (const float*)d_in[28];
  const float* fc2_b  = (const float*)d_in[29];

  float* wsf = (float*)d_ws;
  const size_t MF = (size_t)M_ * 512;     // 8,388,608 floats
  float* buf0 = wsf + 0 * MF;             // sc        -> later q
  float* buf1 = wsf + 1 * MF;             // h1/h2(2x) -> later v; att at +MF
  float* buf2 = wsf + 3 * MF;             // h3        -> later k
  float* buf3 = wsf + 4 * MF;             // x2 (live until gated residual)
  float* buf4 = wsf + 5 * MF;             // amt       -> later fc1 hidden
  float* buf5 = wsf + 6 * MF;             // h0 / concat(2x) -> later fused out
  float* buf6 = wsf + 8 * MF;             // g         -> later rg
  float* buf7 = wsf + 9 * MF;             // out2 (post-norm)
  // total: 10*MF floats = 320 MB of d_ws

  const dim3 blk(256);
  auto gemm = [&](const float* A, const float* W, const float* bias, float* Y,
                  int K, int N, int act) {
    dim3 grid(N / 64, M_ / 128);
    gemm_wmma_nt<<<grid, blk, 0, stream>>>(A, W, bias, Y, M_, N, K, act);
  };

  // --- FeatureEnhancementLayer ---
  gemm(x, fe_wsc, fe_bsc, buf0, 512, 512, 0);                       // sc
  layernorm_kernel<<<M_, 256, 0, stream>>>(x, ln1_g, ln1_b, buf5, 512, 0);
  gemm(buf5, fe_w1, fe_b1, buf1, 512, 1024, 0);                     // h1
  layernorm_kernel<<<M_, 256, 0, stream>>>(buf1, ln2_g, ln2_b, buf1, 1024, 1); // gelu(ln2)
  gemm(buf1, fe_w2, fe_b2, buf2, 1024, 512, 0);                     // h3
  combine_ln_kernel<<<M_, 256, 0, stream>>>(0, buf2, x, buf0, gamma,
                                            ln1_g, ln1_b, buf3);    // x2

  // --- projections ---
  gemm(buf3, wq, nullptr, buf0, 512, 512, 0);                       // q
  gemm(buf3, wk, nullptr, buf2, 512, 512, 0);                       // k
  gemm(buf3, wv, nullptr, buf1, 512, 512, 0);                       // v

  // --- causal attention + rel bias (flash-style, WMMA) ---
  attn_kernel<<<dim3(8, B_ * NH_), 128, 0, stream>>>(buf0, buf2, buf1, relb,
                                                     buf1 + MF);    // att_out
  // --- depthwise conv added in ---
  conv_add_kernel<<<(unsigned)(MF / 256), 256, 0, stream>>>(buf1, conv_w, conv_b,
                                                            buf1 + MF);
  // --- AMT linear-attention scan ---
  amt_scan_kernel<<<B_ * NH_, 256, 0, stream>>>(buf0, buf2, buf1, buf4);

  // --- gated fusion ---
  concat_kernel<<<(unsigned)(2 * MF / 256), 256, 0, stream>>>(buf1 + MF, buf4, buf5);
  gemm(buf5, gate_w, gate_b, buf6, 1024, 512, 1);                   // g = sigmoid
  fuse_kernel<<<(unsigned)(MF / 256), 256, 0, stream>>>(buf6, buf4, buf1 + MF, buf5);

  // --- gated residual + norm ---
  gemm(buf5, rg_w, rg_b, buf6, 512, 512, 1);                        // rg = sigmoid
  combine_ln_kernel<<<M_, 256, 0, stream>>>(1, buf3, buf6, buf5, nullptr,
                                            norm_g, norm_b, buf7);  // out (normed)

  // --- prediction head ---
  gemm(buf7, fc1_w, fc1_b, buf4, 512, 256, 0);
  layernorm_kernel<<<M_, 256, 0, stream>>>(buf4, fcln_g, fcln_b, buf4, 256, 1);
  gemm(buf4, fc2_w, fc2_b, (float*)d_out, 256, 128, 1);             // sigmoid out
}
